// SimplifiedTextGuidedDecomposer_1400159338926
// MI455X (gfx1250) — matrix operand order repaired and checked by reading, not compiled, not run on hardware
//
#include <hip/hip_runtime.h>
#include <hip/hip_bf16.h>
#include <math.h>

typedef __bf16 bf16_t;
typedef __attribute__((ext_vector_type(16))) __bf16 v16bf;
typedef __attribute__((ext_vector_type(8)))  float  v8f;

#define BB 16
#define NN 196
#define DD 512
#define MM 16
#define KK 20
#define BN (BB*NN)           // 3136
#define ROWS (BN*KK)         // 62720
#define MTILES (ROWS/16)     // 3920 (exact)
#define KTOP 51              // int(512*0.1)
#define LN_EPS 1e-5f

// ---- workspace layout (bytes) ----
#define WS_TC   0                          // float [20][512]   centered text feats
#define WS_WT   (WS_TC  + KK*DD*4)         // bf16  [512][512]  W1 transposed (cols contiguous)
#define WS_W2T  (WS_WT  + DD*DD*2)         // bf16  [16][512]   W2 transposed
#define WS_QP   (WS_W2T + MM*DD*2)         // float [3136][16]  Q_patch accumulator
// total = 782336 bytes

union FragU { v16bf v; uint4 q[2]; bf16_t e[16]; };

// ---------------------------------------------------------------------------
// Kernel 0: prep — center text features, bf16-transpose W1/W2, zero Q_patch
// ---------------------------------------------------------------------------
__global__ __launch_bounds__(256) void prep_kernel(
    const float* __restrict__ txt, const float* __restrict__ W1,
    const float* __restrict__ W2,
    float* __restrict__ tc, bf16_t* __restrict__ Wt,
    bf16_t* __restrict__ W2t, float* __restrict__ Qp)
{
  int tid    = blockIdx.x * blockDim.x + threadIdx.x;
  int stride = gridDim.x * blockDim.x;

  if (tid < DD) {                       // tc[k][d] = t[k][d] - mean_k t[.][d]
    float s = 0.f;
    for (int k = 0; k < KK; ++k) s += txt[k*DD + tid];
    float mean = s * (1.0f / KK);
    for (int k = 0; k < KK; ++k) tc[k*DD + tid] = txt[k*DD + tid] - mean;
  }
  for (int i = tid; i < DD*DD; i += stride) {      // Wt[n][k] = W1[k][n]
    int n = i >> 9, k = i & (DD-1);
    Wt[i] = (bf16_t)W1[k*DD + n];
  }
  for (int i = tid; i < MM*DD; i += stride) {      // W2t[m][k] = W2[k][m]
    int m = i >> 9, k = i & (DD-1);
    W2t[i] = (bf16_t)W2[k*MM + m];
  }
  for (int i = tid; i < BN*MM; i += stride) Qp[i] = 0.f;
}

// ---------------------------------------------------------------------------
// Kernel 1: fused (F⊙tc)@W1 -> LN -> GELU -> @W2 -> atomic mean into Q_patch
// One workgroup = one 16-row tile of the 62720-row GEMM. 256 threads / 8 waves.
// ---------------------------------------------------------------------------
__global__ __launch_bounds__(256) void gemm_kernel(
    const float* __restrict__ F, const float* __restrict__ b1,
    const float* __restrict__ ln_g, const float* __restrict__ ln_b,
    const float* __restrict__ b2,
    const float* __restrict__ tc, const bf16_t* __restrict__ Wt,
    const bf16_t* __restrict__ W2t, float* __restrict__ Qp)
{
  // 48 KB LDS, two overlaid regions:
  //  region A (16 KB): A-tile bf16[16][512], later reused as H2 bf16[16][512]
  //  region B (32 KB): h fp32[16][512], first 8 KB later reused as Hp[8][16][16]
  __shared__ float smem[12288];
  bf16_t* A_lds = (bf16_t*)smem;           // [16][512] bf16
  bf16_t* H2    = A_lds;                   // reuse after phase 2
  float*  Hl    = smem + 4096;             // [16][512] f32
  float*  Hp    = Hl;                      // [8][16][16] f32 partials (reuse)

  const int t    = threadIdx.x;
  const int wv   = t >> 5;                 // wave 0..7
  const int lane = t & 31;
  const int half = lane >> 4;              // 0: lanes 0-15, 1: lanes 16-31
  const int l15  = lane & 15;
  const int tile = blockIdx.x;

  // ---- phase 1: build A tile in LDS: A[r][k] = F[bn(r),k] * tc[kk(r),k] ----
  for (int i = t; i < 16*DD; i += 256) {
    int r  = i >> 9;
    int k  = i & (DD-1);
    int gr = tile*16 + r;
    int bn = gr / KK;
    int kk = gr - bn*KK;
    A_lds[i] = (bf16_t)(F[(size_t)bn*DD + k] * tc[kk*DD + k]);
  }
  __syncthreads();

  // ---- phase 2: h = A @ W1 ----
  // Each wave owns 4 adjacent 16-col tiles (64 columns). ks-outer loop: one
  // A-fragment LDS load feeds 4 WMMAs (4x less DS traffic than ct-outer).
  {
    v8f acc0 = {}, acc1 = {}, acc2 = {}, acc3 = {};
    const bf16_t* bcol0 = Wt + (size_t)(wv*64 +  0 + l15) * DD;
    const bf16_t* bcol1 = Wt + (size_t)(wv*64 + 16 + l15) * DD;
    const bf16_t* bcol2 = Wt + (size_t)(wv*64 + 32 + l15) * DD;
    const bf16_t* bcol3 = Wt + (size_t)(wv*64 + 48 + l15) * DD;
    for (int ks = 0; ks < 16; ++ks) {
      // A frag: lane half selects K {0..15}/{16..31} (matches B)
      FragU af;
      int ac = ks*32 + half*16;
      af.q[0] = *(const uint4*)&A_lds[l15*DD + ac];
      af.q[1] = *(const uint4*)&A_lds[l15*DD + ac + 8];
      int bo = ks*32 + half*16;            // B frag: 16 consecutive K per lane
      FragU b0, b1f, b2f, b3f;
      const uint4* p0 = (const uint4*)(bcol0 + bo);
      const uint4* p1 = (const uint4*)(bcol1 + bo);
      const uint4* p2 = (const uint4*)(bcol2 + bo);
      const uint4* p3 = (const uint4*)(bcol3 + bo);
      __builtin_prefetch(p0 + 8, 0, 1);
      __builtin_prefetch(p2 + 8, 0, 1);
      b0.q[0]  = p0[0]; b0.q[1]  = p0[1];
      b1f.q[0] = p1[0]; b1f.q[1] = p1[1];
      b2f.q[0] = p2[0]; b2f.q[1] = p2[1];
      b3f.q[0] = p3[0]; b3f.q[1] = p3[1];
      acc0 = __builtin_amdgcn_wmma_f32_16x16x32_bf16(false, af.v, false, b0.v,  (short)0, acc0, false, false);
      acc1 = __builtin_amdgcn_wmma_f32_16x16x32_bf16(false, af.v, false, b1f.v, (short)0, acc1, false, false);
      acc2 = __builtin_amdgcn_wmma_f32_16x16x32_bf16(false, af.v, false, b2f.v, (short)0, acc2, false, false);
      acc3 = __builtin_amdgcn_wmma_f32_16x16x32_bf16(false, af.v, false, b3f.v, (short)0, acc3, false, false);
    }
    // C layout: VGPR i -> row i + half*8, col l15
    int row0 = half*8;
    float bv0 = b1[wv*64 +  0 + l15];
    float bv1 = b1[wv*64 + 16 + l15];
    float bv2 = b1[wv*64 + 32 + l15];
    float bv3 = b1[wv*64 + 48 + l15];
    for (int i = 0; i < 8; ++i) {
      Hl[(i + row0)*DD + wv*64 +  0 + l15] = acc0[i] + bv0;
      Hl[(i + row0)*DD + wv*64 + 16 + l15] = acc1[i] + bv1;
      Hl[(i + row0)*DD + wv*64 + 32 + l15] = acc2[i] + bv2;
      Hl[(i + row0)*DD + wv*64 + 48 + l15] = acc3[i] + bv3;
    }
  }
  __syncthreads();

  // ---- phase 3: LayerNorm + exact GELU per row, write bf16 into H2 ----
  for (int rr = 0; rr < 2; ++rr) {
    int r = wv*2 + rr;
    float s = 0.f, s2 = 0.f;
    for (int i = 0; i < 16; ++i) {
      float x = Hl[r*DD + i*32 + lane];
      s += x; s2 += x*x;
    }
    for (int off = 16; off > 0; off >>= 1) {
      s  += __shfl_xor(s,  off, 32);
      s2 += __shfl_xor(s2, off, 32);
    }
    float mu   = s * (1.0f/DD);
    float var  = s2 * (1.0f/DD) - mu*mu;
    float rstd = rsqrtf(var + LN_EPS);
    for (int i = 0; i < 16; ++i) {
      int d = i*32 + lane;
      float x  = (Hl[r*DD + d] - mu) * rstd * ln_g[d] + ln_b[d];
      float ge = 0.5f * x * (1.0f + erff(x * 0.70710678118f));
      H2[r*DD + d] = (bf16_t)ge;
    }
  }
  __syncthreads();

  // ---- phase 4: comp = h @ W2 (K split across 8 waves, 2 k-steps each) ----
  {
    v8f acc = {};
    const bf16_t* bcol = W2t + (size_t)l15 * DD;
    for (int kc = 0; kc < 2; ++kc) {
      int ks = wv*2 + kc;
      FragU af, bf_;
      int ac = ks*32 + half*16;
      af.q[0] = *(const uint4*)&H2[l15*DD + ac];
      af.q[1] = *(const uint4*)&H2[l15*DD + ac + 8];
      const uint4* bp = (const uint4*)(bcol + ks*32 + half*16);
      bf_.q[0] = bp[0];
      bf_.q[1] = bp[1];
      acc = __builtin_amdgcn_wmma_f32_16x16x32_bf16(
          false, af.v, false, bf_.v, (short)0, acc, false, false);
    }
    for (int i = 0; i < 8; ++i)
      Hp[(wv*16 + i + half*8)*16 + l15] = acc[i];
  }
  __syncthreads();

  // ---- phase 5: reduce partials, add b2, atomic mean-over-K into Q_patch ----
  {
    int r = t >> 4, m = t & 15;
    float s = 0.f;
    for (int w = 0; w < 8; ++w) s += Hp[(w*16 + r)*16 + m];
    int gr = tile*16 + r;
    int bn = gr / KK;
    atomicAdd(&Qp[bn*MM + m], (s + b2[m]) * (1.0f/KK));
  }
}

// ---------------------------------------------------------------------------
// Kernel 2: Q = F ⊙ templatesᵀ ⊙ Q_patch with exact top-51-of-512 |.| mask.
// One workgroup per (b,n). Threshold = 51st largest via uint-bit binary search
// (monotone for non-negative floats; matches reference's A >= thresh ties).
// ---------------------------------------------------------------------------
__global__ __launch_bounds__(256) void topk_kernel(
    const float* __restrict__ F, const float* __restrict__ templates,
    const float* __restrict__ Qp, float* __restrict__ out)
{
  __shared__ float    Tl[DD*MM];   // templates transposed: Tl[d][m]
  __shared__ float    Fl[DD];
  __shared__ float    qpl[MM];
  __shared__ unsigned thr[MM];

  const int t    = threadIdx.x;
  const int wv   = t >> 5;
  const int lane = t & 31;
  const int bn   = blockIdx.x;

  for (int i = t; i < MM*DD; i += 256) {   // coalesced read, transposed store
    int m = i >> 9, d = i & (DD-1);
    Tl[d*MM + m] = templates[i];
  }
  for (int i = t; i < DD; i += 256) Fl[i] = F[(size_t)bn*DD + i];
  if (t < MM) qpl[t] = Qp[bn*MM + t];
  __syncthreads();

  // threshold search: wave wv handles m = 2*wv, 2*wv+1
  for (int rr = 0; rr < 2; ++rr) {
    int m = wv*2 + rr;
    float qpv = qpl[m];
    unsigned bits[16];
    for (int i = 0; i < 16; ++i) {
      int d = i*32 + lane;
      float q = Fl[d] * Tl[d*MM + m] * qpv;
      bits[i] = __float_as_uint(fabsf(q));
    }
    unsigned lo = 0u, hi = 0x7F800000u;    // [0, +inf)
    while (lo < hi) {                      // largest u with count(|q|>=u) >= 51
      unsigned mid = lo + ((hi - lo + 1u) >> 1);
      int c = 0;
      for (int i = 0; i < 16; ++i) c += (bits[i] >= mid);
      for (int off = 16; off > 0; off >>= 1) c += __shfl_xor(c, off, 32);
      if (c >= KTOP) lo = mid; else hi = mid - 1u;
    }
    if (lane == 0) thr[m] = lo;
  }
  __syncthreads();

  // masked write, fully coalesced (m fastest)
  {
    int m = t & 15;
    unsigned tm = thr[m];
    float qpv = qpl[m];
    for (int it = 0; it < 32; ++it) {
      int d = it*16 + (t >> 4);
      float q = Fl[d] * Tl[d*MM + m] * qpv;
      bool keep = __float_as_uint(fabsf(q)) >= tm;
      out[((size_t)bn*DD + d)*MM + m] = keep ? q : 0.f;
    }
  }
}

// ---------------------------------------------------------------------------
extern "C" void kernel_launch(void* const* d_in, const int* in_sizes, int n_in,
                              void* d_out, int out_size, void* d_ws, size_t ws_size,
                              hipStream_t stream) {
  (void)in_sizes; (void)n_in; (void)out_size; (void)ws_size;
  const float* F    = (const float*)d_in[0];
  const float* txt  = (const float*)d_in[1];
  const float* W1   = (const float*)d_in[2];
  const float* b1   = (const float*)d_in[3];
  const float* ln_g = (const float*)d_in[4];
  const float* ln_b = (const float*)d_in[5];
  const float* W2   = (const float*)d_in[6];
  const float* b2   = (const float*)d_in[7];
  const float* tmpl = (const float*)d_in[8];

  char* ws = (char*)d_ws;
  float*  tc  = (float*)(ws + WS_TC);
  bf16_t* Wt  = (bf16_t*)(ws + WS_WT);
  bf16_t* W2t = (bf16_t*)(ws + WS_W2T);
  float*  Qp  = (float*)(ws + WS_QP);

  prep_kernel<<<512, 256, 0, stream>>>(txt, W1, W2, tc, Wt, W2t, Qp);
  gemm_kernel<<<MTILES, 256, 0, stream>>>(F, b1, ln_g, ln_b, b2, tc, Wt, W2t, Qp);
  topk_kernel<<<BN, 256, 0, stream>>>(F, tmpl, Qp, (float*)d_out);
}
